// GraphCastNet_49435073577063
// MI455X (gfx1250) — matrix operand order repaired and checked by reading, not compiled, not run on hardware
//
#include <hip/hip_runtime.h>
#include <hip/hip_bf16.h>

// ---------------- problem constants ----------------
#define HH 121
#define WW 240
#define NG (HH*WW)          // 29040 grid nodes
#define NM 2562             // mesh nodes
#define EMM 20460           // mesh-mesh edges
#define EG2M (3*NG)         // 87120 grid->mesh edges
#define EM2G (3*NG)         // 87120 mesh->grid edges
#define CIN 474
#define CINP 480            // padded to multiple of 32
#define COUT 227
#define D 256
#define NLAYERS 4

typedef __attribute__((ext_vector_type(16))) __bf16 bf16x16;
typedef __attribute__((ext_vector_type(8)))  float  floatx8;

union FragB { bf16x16 v; uint4 q[2]; };

__device__ __forceinline__ unsigned short f2bf(float f) {
    unsigned int x = __float_as_uint(f);
    unsigned int r = x + 0x7FFFu + ((x >> 16) & 1u);   // round-to-nearest-even
    return (unsigned short)(r >> 16);
}

// ---------------------------------------------------------------------------
// Fused MLP: Y = [LayerNorm](SiLU(X@W1+b1)@W2+b2)[*g+bt] [+res]
// X: bf16 [N, din_p] row-major (din_p multiple of 32)
// W1sw: bf16 swizzled [(din_p/32)][256 cols][2 halves][16]  (B-fragment order)
// W2sw: bf16 swizzled [(256/32)][256 cols][2][16], cols >= dout zero-padded
// Yf: fp32 [N, dout]; Ybf: bf16 [N, 256] (nullable); res: fp32 [N, dout] (nullable)
// Block = 256 threads (8 wave32), tile = 32 rows x 256 cols.
// ---------------------------------------------------------------------------
__global__ __launch_bounds__(256)
void mlp_kernel(const unsigned short* __restrict__ X,
                const unsigned short* __restrict__ W1,
                const float* __restrict__ b1,
                const unsigned short* __restrict__ W2,
                const float* __restrict__ b2,
                const float* __restrict__ gamma,
                const float* __restrict__ beta,
                const float* __restrict__ res,
                float* __restrict__ Yf,
                unsigned short* __restrict__ Ybf,
                int N, int din_p, int dout)
{
    __shared__ unsigned short Hs[32 * 256];   // stage-1 activations (bf16)
    __shared__ float          Ys[32 * 256];   // stage-2 raw output (fp32)
    __shared__ float          Ps[256], Ps2[256];
    __shared__ float          Mu[32], Rstd[32];

    const int tid  = threadIdx.x;
    const int lane = tid & 31;
    const int wave = tid >> 5;
    const int wrow = wave >> 2;     // 0..1 : 16-row group
    const int wcol = wave & 3;      // 0..3 : 64-col group
    const int row0 = blockIdx.x * 32;
    const int l16  = lane & 15;
    const int lh   = lane >> 4;     // lane-half (K interleave select)
    const int k0   = lh * 8;

    // ---------------- stage 1: H = SiLU(X @ W1 + b1) ----------------
    floatx8 acc[4];
    #pragma unroll
    for (int t = 0; t < 4; t++)
        #pragma unroll
        for (int i = 0; i < 8; i++) acc[t][i] = 0.f;

    int arow = row0 + wrow * 16 + l16;
    if (arow >= N) arow = N - 1;               // clamp (stores are guarded later)
    const unsigned short* Xrow = X + (size_t)arow * din_p;

    // per-lane base into the swizzled weight layout (col = wcol*64 + t*16 + l16)
    const size_t wlane = ((size_t)(wcol * 64 + l16) * 2 + lh) * 16;

    const int ksteps = din_p >> 5;
    for (int ks = 0; ks < ksteps; ks++) {
        FragB A;
        const uint4* ap = (const uint4*)(Xrow + ks * 32 + k0);
        A.q[0] = ap[0];        // K = 32ks + k0 + 0..7
        A.q[1] = ap[2];        // K = 32ks + k0 + 16..23

        // batch all four B-tile loads so they issue as one clause
        FragB Bf[4];
        const unsigned short* wbase = W1 + (size_t)ks * 8192 + wlane;
        #pragma unroll
        for (int t = 0; t < 4; t++) {
            const uint4* bp = (const uint4*)(wbase + t * 512);   // t*16 cols * 32 elems
            Bf[t].q[0] = bp[0];
            Bf[t].q[1] = bp[1];
        }
        #pragma unroll
        for (int t = 0; t < 4; t++)
            acc[t] = __builtin_amdgcn_wmma_f32_16x16x32_bf16(
                false, A.v, false, Bf[t].v, (short)0, acc[t], false, false);
    }

    #pragma unroll
    for (int t = 0; t < 4; t++) {
        const int c = wcol * 64 + t * 16 + l16;
        const float bb = b1[c];
        #pragma unroll
        for (int r = 0; r < 8; r++) {
            float x = acc[t][r] + bb;
            float s = x / (1.f + __expf(-x));          // SiLU
            const int rl = wrow * 16 + lh * 8 + r;     // C/D layout row
            Hs[rl * 256 + c] = f2bf(s);
        }
    }
    __syncthreads();

    // ---------------- stage 2: Y = H @ W2 + b2 ----------------
    floatx8 acc2[4];
    #pragma unroll
    for (int t = 0; t < 4; t++)
        #pragma unroll
        for (int i = 0; i < 8; i++) acc2[t][i] = 0.f;

    const int hrow = wrow * 16 + l16;
    #pragma unroll
    for (int ks = 0; ks < 8; ks++) {
        FragB A;
        const uint4* ap = (const uint4*)(&Hs[hrow * 256 + ks * 32 + k0]);
        A.q[0] = ap[0];
        A.q[1] = ap[2];

        FragB Bf[4];
        const unsigned short* wbase = W2 + (size_t)ks * 8192 + wlane;
        #pragma unroll
        for (int t = 0; t < 4; t++) {
            const uint4* bp = (const uint4*)(wbase + t * 512);
            Bf[t].q[0] = bp[0];
            Bf[t].q[1] = bp[1];
        }
        #pragma unroll
        for (int t = 0; t < 4; t++)
            acc2[t] = __builtin_amdgcn_wmma_f32_16x16x32_bf16(
                false, A.v, false, Bf[t].v, (short)0, acc2[t], false, false);
    }

    #pragma unroll
    for (int t = 0; t < 4; t++) {
        const int c = wcol * 64 + t * 16 + l16;
        const float bb = (c < dout) ? b2[c] : 0.f;
        #pragma unroll
        for (int r = 0; r < 8; r++) {
            const int rl = wrow * 16 + lh * 8 + r;
            Ys[rl * 256 + c] = acc2[t][r] + bb;
        }
    }
    __syncthreads();

    // ---------------- LayerNorm stats: 8 threads per row, 2-level reduce ----
    if (gamma) {
        const int rown = tid >> 3;      // 0..31
        const int sub  = tid & 7;       // 0..7 -> 32 cols each
        float s = 0.f, s2 = 0.f;
        const int cbase = sub * 32;
        #pragma unroll 8
        for (int c = cbase; c < cbase + 32; c++) {
            if (c < dout) {
                float v = Ys[rown * 256 + c];
                s += v; s2 += v * v;
            }
        }
        Ps[tid] = s; Ps2[tid] = s2;
    }
    __syncthreads();
    if (gamma && tid < 32) {
        float s = 0.f, s2 = 0.f;
        #pragma unroll
        for (int k = 0; k < 8; k++) { s += Ps[tid * 8 + k]; s2 += Ps2[tid * 8 + k]; }
        const float inv = 1.f / (float)dout;
        const float mu = s * inv;
        const float var = s2 * inv - mu * mu;
        Mu[tid] = mu;
        Rstd[tid] = rsqrtf(var + 1e-5f);
    }
    __syncthreads();

    // ---------------- epilogue: norm + residual + dual-precision store -----
    for (int i = tid; i < 32 * dout; i += 256) {
        const int rl = i / dout;
        const int c  = i - rl * dout;
        const int grow = row0 + rl;
        if (grow >= N) continue;
        float v = Ys[rl * 256 + c];
        if (gamma) v = (v - Mu[rl]) * Rstd[rl] * gamma[c] + beta[c];
        if (res)   v += res[(size_t)grow * dout + c];
        Yf[(size_t)grow * dout + c] = v;
        if (Ybf) Ybf[(size_t)grow * 256 + c] = f2bf(v);
    }
}

// ---------------------------------------------------------------------------
// Weight fp32 -> bf16 with B-fragment swizzle:
// dst[((B*256 + c)*2 + h)*16 + j] = W[(32B + 8h + j + (j & 8))*cols + c]   (0 if OOB)
// ---------------------------------------------------------------------------
__global__ void w_swizzle_kernel(const float* __restrict__ W, unsigned short* __restrict__ dst,
                                 int rows, int rows_p, int cols)
{
    const int tid = blockIdx.x * 256 + threadIdx.x;
    const int total = (rows_p >> 5) * 8192;
    if (tid >= total) return;
    const int j = tid & 15;
    const int h = (tid >> 4) & 1;
    const int c = (tid >> 5) & 255;
    const int B = tid >> 13;
    const int k = B * 32 + h * 8 + j + (j & 8);
    float v = (k < rows && c < cols) ? W[(size_t)k * cols + c] : 0.f;
    dst[tid] = f2bf(v);
}

// (1, CIN, H, W) fp32 -> [NG, CINP] bf16 (transposed, zero-padded)
__global__ void transpose_grid_kernel(const float* __restrict__ g, unsigned short* __restrict__ X0)
{
    const int tid = blockIdx.x * 256 + threadIdx.x;
    if (tid >= NG * CINP) return;
    const int n = tid / CINP;
    const int c = tid - n * CINP;
    float v = (c < CIN) ? g[(size_t)c * NG + n] : 0.f;
    X0[tid] = f2bf(v);
}

// [N, cin] fp32 -> [N, 32] bf16 zero-padded
__global__ void pad_bf16_kernel(const float* __restrict__ src, unsigned short* __restrict__ dst,
                                int N, int cin)
{
    const int tid = blockIdx.x * 256 + threadIdx.x;
    if (tid >= N * 32) return;
    const int n = tid >> 5;
    const int c = tid & 31;
    float v = (c < cin) ? src[(size_t)n * cin + c] : 0.f;
    dst[tid] = f2bf(v);
}

// cat[e] = [eF[e] | sF[src[e]] | dF[dst[e]]], all bf16 stride 256 -> stride 768
__global__ void gather_cat_edge_kernel(const unsigned short* __restrict__ eF,
                                       const unsigned short* __restrict__ sF,
                                       const unsigned short* __restrict__ dF,
                                       const int* __restrict__ src,
                                       const int* __restrict__ dst,
                                       unsigned short* __restrict__ cat, int E)
{
    const int tid = blockIdx.x * 256 + threadIdx.x;
    if (tid >= E * 96) return;
    const int e  = tid / 96;
    const int ch = tid - e * 96;
    uint4 v;
    if (ch < 32)
        v = ((const uint4*)(eF + (size_t)e * 256))[ch];
    else if (ch < 64)
        v = ((const uint4*)(sF + (size_t)src[e] * 256))[ch - 32];
    else
        v = ((const uint4*)(dF + (size_t)dst[e] * 256))[ch - 64];
    ((uint4*)(cat + (size_t)e * 768))[ch] = v;
}

// cat[n] = [bf16(agg[n]) | nodeF[n]] -> stride 512
__global__ void gather_cat_node_kernel(const float* __restrict__ agg,
                                       const unsigned short* __restrict__ nodeF,
                                       unsigned short* __restrict__ cat, int Nn)
{
    const int tid = blockIdx.x * 256 + threadIdx.x;
    if (tid >= Nn * 64) return;
    const int n  = tid >> 6;
    const int ch = tid & 63;
    if (ch < 32) {
        const float4* a = (const float4*)(agg + (size_t)n * 256) + ch * 2;
        float4 a0 = a[0], a1 = a[1];
        union { uint4 q; unsigned short u[8]; } o;
        o.u[0] = f2bf(a0.x); o.u[1] = f2bf(a0.y); o.u[2] = f2bf(a0.z); o.u[3] = f2bf(a0.w);
        o.u[4] = f2bf(a1.x); o.u[5] = f2bf(a1.y); o.u[6] = f2bf(a1.z); o.u[7] = f2bf(a1.w);
        ((uint4*)(cat + (size_t)n * 512))[ch] = o.q;
    } else {
        ((uint4*)(cat + (size_t)n * 512))[ch] = ((const uint4*)(nodeF + (size_t)n * 256))[ch - 32];
    }
}

__global__ void zero_f32_kernel(float* __restrict__ p, int n)
{
    const int tid = blockIdx.x * 256 + threadIdx.x;
    if (tid < n) p[tid] = 0.f;
}

// segment_sum: agg[dst[e]] += eY[e]   (fp32 atomics; e_Y stride 256)
__global__ void scatter_add_kernel(const float* __restrict__ eY, const int* __restrict__ dst,
                                   float* __restrict__ agg, int E)
{
    const int tid = blockIdx.x * 256 + threadIdx.x;
    if (tid >= E * 256) return;
    const int e = tid >> 8;
    const int c = tid & 255;
    atomicAdd(agg + (size_t)dst[e] * 256 + c, eY[tid]);
}

// [NG, COUT] -> (1, COUT, H, W)
__global__ void out_transpose_kernel(const float* __restrict__ Y, float* __restrict__ out)
{
    const int tid = blockIdx.x * 256 + threadIdx.x;
    if (tid >= COUT * NG) return;
    const int c = tid / NG;
    const int n = tid - c * NG;
    out[tid] = Y[(size_t)n * COUT + c];
}

// ---------------------------------------------------------------------------
extern "C" void kernel_launch(void* const* d_in, const int* in_sizes, int n_in,
                              void* d_out, int out_size, void* d_ws, size_t ws_size,
                              hipStream_t stream)
{
    (void)in_sizes; (void)n_in; (void)out_size; (void)ws_size;

    struct MLPp { const float *W1, *b1, *W2, *b2, *g, *bt; };
    int idx = 0;
    const float* grid_nfeat = (const float*)d_in[idx++];
    auto take = [&](bool norm) {
        MLPp m;
        m.W1 = (const float*)d_in[idx++]; m.b1 = (const float*)d_in[idx++];
        m.W2 = (const float*)d_in[idx++]; m.b2 = (const float*)d_in[idx++];
        if (norm) { m.g = (const float*)d_in[idx++]; m.bt = (const float*)d_in[idx++]; }
        else      { m.g = nullptr; m.bt = nullptr; }
        return m;
    };
    MLPp embed_grid    = take(true);
    MLPp embed_mesh    = take(true);
    MLPp embed_g2m_e   = take(true);
    MLPp embed_mesh_e  = take(true);
    MLPp g2m_edge      = take(true);
    MLPp g2m_node_mesh = take(true);
    MLPp g2m_node_grid = take(true);
    MLPp proc_edge[NLAYERS], proc_node[NLAYERS];
    for (int i = 0; i < NLAYERS; i++) proc_edge[i] = take(true);
    for (int i = 0; i < NLAYERS; i++) proc_node[i] = take(true);
    MLPp embed_m2g_e = take(true);
    MLPp m2g_edge    = take(true);
    MLPp m2g_node    = take(true);
    MLPp final_      = take(false);
    const float* mesh_ndata = (const float*)d_in[idx++];
    const float* g2m_edata  = (const float*)d_in[idx++];
    const float* mesh_edata = (const float*)d_in[idx++];
    const float* m2g_edata  = (const float*)d_in[idx++];
    const int* g2m_src  = (const int*)d_in[idx++];
    const int* g2m_dst  = (const int*)d_in[idx++];
    const int* mesh_src = (const int*)d_in[idx++];
    const int* mesh_dst = (const int*)d_in[idx++];
    const int* m2g_src  = (const int*)d_in[idx++];
    const int* m2g_dst  = (const int*)d_in[idx++];

    // ----- workspace bump allocator -----
    char* wsp = (char*)d_ws;
    size_t off = 0;
    auto alloc = [&](size_t bytes) -> void* {
        void* p = wsp + off;
        off = (off + bytes + 255) & ~(size_t)255;
        return p;
    };

    auto gsz = [](int n) { return (n + 255) / 256; };

    // weight prep: convert+swizzle to bf16
    auto prep = [&](const MLPp& m, int din, int din_p, int dout,
                    unsigned short*& w1sw, unsigned short*& w2sw) {
        w1sw = (unsigned short*)alloc((size_t)din_p * 256 * 2);
        w2sw = (unsigned short*)alloc((size_t)256 * 256 * 2);
        int n1 = (din_p / 32) * 8192;
        hipLaunchKernelGGL(w_swizzle_kernel, dim3(gsz(n1)), dim3(256), 0, stream,
                           m.W1, w1sw, din, din_p, 256);
        hipLaunchKernelGGL(w_swizzle_kernel, dim3(gsz(65536)), dim3(256), 0, stream,
                           m.W2, w2sw, 256, 256, dout);
    };

    unsigned short *sw_eg1, *sw_eg2;   prep(embed_grid,    CIN, CINP, 256, sw_eg1, sw_eg2);
    unsigned short *sw_em1, *sw_em2;   prep(embed_mesh,    3,   32,   256, sw_em1, sw_em2);
    unsigned short *sw_ge1, *sw_ge2;   prep(embed_g2m_e,   4,   32,   256, sw_ge1, sw_ge2);
    unsigned short *sw_me1, *sw_me2;   prep(embed_mesh_e,  4,   32,   256, sw_me1, sw_me2);
    unsigned short *sw_gE1, *sw_gE2;   prep(g2m_edge,      768, 768,  256, sw_gE1, sw_gE2);
    unsigned short *sw_gNm1, *sw_gNm2; prep(g2m_node_mesh, 512, 512,  256, sw_gNm1, sw_gNm2);
    unsigned short *sw_gNg1, *sw_gNg2; prep(g2m_node_grid, 256, 256,  256, sw_gNg1, sw_gNg2);
    unsigned short *sw_pe1[NLAYERS], *sw_pe2[NLAYERS], *sw_pn1[NLAYERS], *sw_pn2[NLAYERS];
    for (int i = 0; i < NLAYERS; i++) prep(proc_edge[i], 768, 768, 256, sw_pe1[i], sw_pe2[i]);
    for (int i = 0; i < NLAYERS; i++) prep(proc_node[i], 512, 512, 256, sw_pn1[i], sw_pn2[i]);
    unsigned short *sw_m2e1, *sw_m2e2; prep(embed_m2g_e, 4,   32,  256, sw_m2e1, sw_m2e2);
    unsigned short *sw_mE1, *sw_mE2;   prep(m2g_edge,    768, 768, 256, sw_mE1, sw_mE2);
    unsigned short *sw_mN1, *sw_mN2;   prep(m2g_node,    512, 512, 256, sw_mN1, sw_mN2);
    unsigned short *sw_f1, *sw_f2;     prep(final_,      256, 256, COUT, sw_f1, sw_f2);

    // ----- activation buffers -----
    unsigned short* X0     = (unsigned short*)alloc((size_t)NG * CINP * 2);
    unsigned short* padbuf = (unsigned short*)alloc((size_t)EG2M * 32 * 2);
    float*          g_f    = (float*)alloc((size_t)NG * 256 * 4);
    unsigned short* g_bf   = (unsigned short*)alloc((size_t)NG * 256 * 2);
    float*          m_f    = (float*)alloc((size_t)NM * 256 * 4);
    unsigned short* m_bf   = (unsigned short*)alloc((size_t)NM * 256 * 2);
    float*          e_f    = (float*)alloc((size_t)EG2M * 256 * 4);
    unsigned short* e_bf   = (unsigned short*)alloc((size_t)EG2M * 256 * 2);
    unsigned short* catE   = (unsigned short*)alloc((size_t)EG2M * 768 * 2);
    unsigned short* catN   = (unsigned short*)alloc((size_t)NG * 512 * 2);
    float*          agg    = (float*)alloc((size_t)NG * 256 * 4);
    float*          Yfin   = (float*)alloc((size_t)NG * COUT * 4);

    auto run_mlp = [&](const unsigned short* X, unsigned short* w1, const MLPp& m,
                       unsigned short* w2, const float* res, float* Yf,
                       unsigned short* Ybf, int N, int din_p, int dout) {
        hipLaunchKernelGGL(mlp_kernel, dim3((N + 31) / 32), dim3(256), 0, stream,
                           X, w1, m.b1, w2, m.b2, m.g, m.bt, res, Yf, Ybf, N, din_p, dout);
    };

    // ===== encoder embedders =====
    hipLaunchKernelGGL(transpose_grid_kernel, dim3(gsz(NG * CINP)), dim3(256), 0, stream,
                       grid_nfeat, X0);
    run_mlp(X0, sw_eg1, embed_grid, sw_eg2, nullptr, g_f, g_bf, NG, CINP, 256);

    hipLaunchKernelGGL(pad_bf16_kernel, dim3(gsz(NM * 32)), dim3(256), 0, stream,
                       mesh_ndata, padbuf, NM, 3);
    run_mlp(padbuf, sw_em1, embed_mesh, sw_em2, nullptr, m_f, m_bf, NM, 32, 256);

    hipLaunchKernelGGL(pad_bf16_kernel, dim3(gsz(EG2M * 32)), dim3(256), 0, stream,
                       g2m_edata, padbuf, EG2M, 4);
    run_mlp(padbuf, sw_ge1, embed_g2m_e, sw_ge2, nullptr, e_f, e_bf, EG2M, 32, 256);

    // ===== grid2mesh =====
    hipLaunchKernelGGL(gather_cat_edge_kernel, dim3(gsz(EG2M * 96)), dim3(256), 0, stream,
                       e_bf, g_bf, m_bf, g2m_src, g2m_dst, catE, EG2M);
    run_mlp(catE, sw_gE1, g2m_edge, sw_gE2, e_f, e_f, e_bf, EG2M, 768, 256);
    hipLaunchKernelGGL(zero_f32_kernel, dim3(gsz(NM * 256)), dim3(256), 0, stream, agg, NM * 256);
    hipLaunchKernelGGL(scatter_add_kernel, dim3(gsz(EG2M * 256)), dim3(256), 0, stream,
                       e_f, g2m_dst, agg, EG2M);
    hipLaunchKernelGGL(gather_cat_node_kernel, dim3(gsz(NM * 64)), dim3(256), 0, stream,
                       agg, m_bf, catN, NM);
    run_mlp(catN, sw_gNm1, g2m_node_mesh, sw_gNm2, m_f, m_f, m_bf, NM, 512, 256);
    run_mlp(g_bf, sw_gNg1, g2m_node_grid, sw_gNg2, g_f, g_f, g_bf, NG, 256, 256);

    // ===== mesh processor =====
    hipLaunchKernelGGL(pad_bf16_kernel, dim3(gsz(EMM * 32)), dim3(256), 0, stream,
                       mesh_edata, padbuf, EMM, 4);
    run_mlp(padbuf, sw_me1, embed_mesh_e, sw_me2, nullptr, e_f, e_bf, EMM, 32, 256);
    for (int i = 0; i < NLAYERS; i++) {
        hipLaunchKernelGGL(gather_cat_edge_kernel, dim3(gsz(EMM * 96)), dim3(256), 0, stream,
                           e_bf, m_bf, m_bf, mesh_src, mesh_dst, catE, EMM);
        run_mlp(catE, sw_pe1[i], proc_edge[i], sw_pe2[i], e_f, e_f, e_bf, EMM, 768, 256);
        hipLaunchKernelGGL(zero_f32_kernel, dim3(gsz(NM * 256)), dim3(256), 0, stream, agg, NM * 256);
        hipLaunchKernelGGL(scatter_add_kernel, dim3(gsz(EMM * 256)), dim3(256), 0, stream,
                           e_f, mesh_dst, agg, EMM);
        hipLaunchKernelGGL(gather_cat_node_kernel, dim3(gsz(NM * 64)), dim3(256), 0, stream,
                           agg, m_bf, catN, NM);
        run_mlp(catN, sw_pn1[i], proc_node[i], sw_pn2[i], m_f, m_f, m_bf, NM, 512, 256);
    }

    // ===== mesh2grid =====
    hipLaunchKernelGGL(pad_bf16_kernel, dim3(gsz(EM2G * 32)), dim3(256), 0, stream,
                       m2g_edata, padbuf, EM2G, 4);
    run_mlp(padbuf, sw_m2e1, embed_m2g_e, sw_m2e2, nullptr, e_f, e_bf, EM2G, 32, 256);
    hipLaunchKernelGGL(gather_cat_edge_kernel, dim3(gsz(EM2G * 96)), dim3(256), 0, stream,
                       e_bf, m_bf, g_bf, m2g_src, m2g_dst, catE, EM2G);
    run_mlp(catE, sw_mE1, m2g_edge, sw_mE2, e_f, e_f, e_bf, EM2G, 768, 256);
    hipLaunchKernelGGL(zero_f32_kernel, dim3(gsz(NG * 256)), dim3(256), 0, stream, agg, NG * 256);
    hipLaunchKernelGGL(scatter_add_kernel, dim3(gsz(EM2G * 256)), dim3(256), 0, stream,
                       e_f, m2g_dst, agg, EM2G);
    hipLaunchKernelGGL(gather_cat_node_kernel, dim3(gsz(NG * 64)), dim3(256), 0, stream,
                       agg, g_bf, catN, NG);
    run_mlp(catN, sw_mN1, m2g_node, sw_mN2, g_f, g_f, g_bf, NG, 512, 256);

    // ===== final head + output layout =====
    run_mlp(g_bf, sw_f1, final_, sw_f2, nullptr, Yfin, nullptr, NG, 256, COUT);
    hipLaunchKernelGGL(out_transpose_kernel, dim3(gsz(COUT * NG)), dim3(256), 0, stream,
                       Yfin, (float*)d_out);
}
